// IPAdapterFluxAttnProcessor2_0_54434415509677
// MI455X (gfx1250) — compile-verified, matrix operand order checked by reading, not run on hardware
//
#include <hip/hip_runtime.h>
#include <hip/hip_bf16.h>

// ---------------------------------------------------------------------------
// IP-Adapter Flux attention for MI455X (gfx1250, wave32, WMMA bf16).
// All GEMMs and both SDPAs run on v_wmma_f32_16x16x32_bf16 with fp32 accum.
// ---------------------------------------------------------------------------

typedef __attribute__((ext_vector_type(16))) __bf16 v16bf;
typedef __attribute__((ext_vector_type(8)))  __bf16 v8bf;
typedef __attribute__((ext_vector_type(8)))  float  v8f;

#define NH     24
#define HD     128
#define DMODEL 3072
#define STXT   512
#define SIMG   2048
#define SALL   2560
#define NIP    64
#define EMB    4096

// Scheduling fence: only SALU (0x4) and VMEM reads (0x20) may cross, so
// loads keep flowing through WMMA bursts but VALU cvts cannot be sunk into
// them (which would trip the WMMA->VALU WAR hazard and force v_nop padding).
#define SCHED_FENCE() __builtin_amdgcn_sched_barrier(0x24)

__device__ __forceinline__ __bf16 f2bf(float f) {
  // lowers to v_cvt_pk_bf16_f32 on gfx1250
  return (__bf16)f;
}

__device__ __forceinline__ v8f wmma_bf16(v16bf a, v16bf b, v8f c) {
  // (neg_a, A, neg_b, B, c_mod, C, reuse_a, reuse_b)
  return __builtin_amdgcn_wmma_f32_16x16x32_bf16(false, a, false, b, (short)0, c,
                                                 false, false);
}

// 16x32 bf16 A/B fragment from row-major bf16 [rows x ld].
// ISA 16-bit A layout: lane<16 -> row=lane, K={k0..k0+7, k0+16..k0+23};
// lane>=16 -> row=lane-16, K={k0+8..k0+15, k0+24..k0+31}.  B (32x16) from a
// row-major (N,K) source uses the exact same pattern (column = lane&15).
__device__ __forceinline__ v16bf load_frag_bf16(const __bf16* __restrict__ src,
                                                int ld, int row0, int k0) {
  int lane = threadIdx.x & 31;
  const __bf16* p = src + (size_t)(row0 + (lane & 15)) * ld + (k0 + ((lane >> 4) << 3));
  v8bf lo = *reinterpret_cast<const v8bf*>(p);
  v8bf hi = *reinterpret_cast<const v8bf*>(p + 16);
  return __builtin_shufflevector(lo, hi, 0,1,2,3,4,5,6,7,8,9,10,11,12,13,14,15);
}

// Same fragment but streamed from fp32 with in-register bf16 conversion.
__device__ __forceinline__ v16bf load_frag_f32(const float* __restrict__ src,
                                               int ld, int row0, int k0) {
  int lane = threadIdx.x & 31;
  const float* p = src + (size_t)(row0 + (lane & 15)) * ld + (k0 + ((lane >> 4) << 3));
  v16bf r;
#pragma unroll
  for (int i = 0; i < 8; ++i) r[i] = f2bf(p[i]);
#pragma unroll
  for (int i = 0; i < 8; ++i) r[8 + i] = f2bf(p[16 + i]);
  return r;
}

// ---------------------------------------------------------------------------
// GEMM: C[M,N] = A[M,K] * W[N,K]^T + bias[N]   (A,W fp32 in, fp32 out)
// block = 128 threads (4 waves); block tile 128(M) x 64(N); wave tile 32x64.
// All 6 fragments of a k-chunk are materialized, then the 8-WMMA burst is
// fenced so no cvt lands inside the WMMA WAR hazard window.
// ---------------------------------------------------------------------------
__global__ __launch_bounds__(128) void gemm_xwT(
    const float* __restrict__ A, const float* __restrict__ W,
    const float* __restrict__ bias, float* __restrict__ C,
    int M, int N, int K) {
  int wave = threadIdx.x >> 5, lane = threadIdx.x & 31;
  int m0 = blockIdx.y * 128 + wave * 32;
  int n0 = blockIdx.x * 64;
  if (m0 >= M) return;  // small-M tail (IP projections); no barriers below

  v8f acc[2][4];
#pragma unroll
  for (int mi = 0; mi < 2; ++mi)
#pragma unroll
    for (int j = 0; j < 4; ++j)
#pragma unroll
      for (int i = 0; i < 8; ++i) acc[mi][j][i] = 0.0f;

#pragma unroll 2
  for (int k0 = 0; k0 < K; k0 += 32) {
    v16bf a0 = load_frag_f32(A, K, m0,      k0);
    v16bf a1 = load_frag_f32(A, K, m0 + 16, k0);
    v16bf b0 = load_frag_f32(W, K, n0,      k0);
    v16bf b1 = load_frag_f32(W, K, n0 + 16, k0);
    v16bf b2 = load_frag_f32(W, K, n0 + 32, k0);
    v16bf b3 = load_frag_f32(W, K, n0 + 48, k0);
    SCHED_FENCE();
    acc[0][0] = wmma_bf16(a0, b0, acc[0][0]);
    acc[1][0] = wmma_bf16(a1, b0, acc[1][0]);
    acc[0][1] = wmma_bf16(a0, b1, acc[0][1]);
    acc[1][1] = wmma_bf16(a1, b1, acc[1][1]);
    acc[0][2] = wmma_bf16(a0, b2, acc[0][2]);
    acc[1][2] = wmma_bf16(a1, b2, acc[1][2]);
    acc[0][3] = wmma_bf16(a0, b3, acc[0][3]);
    acc[1][3] = wmma_bf16(a1, b3, acc[1][3]);
    SCHED_FENCE();
  }
  // f32 C/D layout: lane L holds rows m = i + 8*(L>>4), col n = L&15
  int cl = lane & 15;
#pragma unroll
  for (int mi = 0; mi < 2; ++mi) {
    int rb = m0 + 16 * mi + ((lane >> 4) << 3);
#pragma unroll
    for (int j = 0; j < 4; ++j) {
      int col = n0 + 16 * j + cl;
      float bv = bias ? bias[col] : 0.0f;
#pragma unroll
      for (int i = 0; i < 8; ++i)
        C[(size_t)(rb + i) * N + col] = acc[mi][j][i] + bv;
    }
  }
}

// ---------------------------------------------------------------------------
// Per-(row, head) RMS norm, in place.  rows<split use w_lo, else w_hi.
// One wave per (row, head); 128 elems -> 4 per lane; shfl_xor tree reduce.
// ---------------------------------------------------------------------------
__global__ __launch_bounds__(256) void rms_heads(
    float* __restrict__ buf, const float* __restrict__ w_lo,
    const float* __restrict__ w_hi, int split_row, int rows) {
  int gw = blockIdx.x * (blockDim.x >> 5) + (threadIdx.x >> 5);
  int lane = threadIdx.x & 31;
  if (gw >= rows * NH) return;
  int row = gw / NH, h = gw % NH;
  float* p = buf + (size_t)row * DMODEL + h * HD;
  float x[4], ss = 0.0f;
#pragma unroll
  for (int i = 0; i < 4; ++i) { x[i] = p[lane + 32 * i]; ss += x[i] * x[i]; }
#pragma unroll
  for (int d = 1; d < 32; d <<= 1) ss += __shfl_xor(ss, d, 32);
  float inv = rsqrtf(ss * (1.0f / 128.0f) + 1e-6f);
  const float* w = (row < split_row) ? w_lo : w_hi;
#pragma unroll
  for (int i = 0; i < 4; ++i) p[lane + 32 * i] = x[i] * inv * w[lane + 32 * i];
}

// ---------------------------------------------------------------------------
// RoPE (pairwise rotate) fp32 -> bf16; optionally also emit pre-RoPE bf16
// for rows >= pre_row0 (used by the IP branch's query).
// ---------------------------------------------------------------------------
__global__ __launch_bounds__(256) void rope_cvt(
    const float* __restrict__ src, const float* __restrict__ cosb,
    const float* __restrict__ sinb, __bf16* __restrict__ dst,
    __bf16* __restrict__ pre, int rows, int pre_row0) {
  size_t idx = (size_t)blockIdx.x * blockDim.x + threadIdx.x;
  size_t total = (size_t)rows * NH * (HD / 2);
  if (idx >= total) return;
  int pair = (int)(idx % (NH * (HD / 2)));
  int row  = (int)(idx / (NH * (HD / 2)));
  int h = pair >> 6, hd0 = (pair & 63) * 2;
  size_t off = (size_t)row * DMODEL + h * HD + hd0;
  float x0 = src[off], x1 = src[off + 1];
  size_t ro = (size_t)row * HD + hd0;
  float c0 = cosb[ro], c1 = cosb[ro + 1];
  float s0 = sinb[ro], s1 = sinb[ro + 1];
  dst[off]     = f2bf(x0 * c0 - x1 * s0);
  dst[off + 1] = f2bf(x1 * c1 + x0 * s1);
  if (pre && row >= pre_row0) {
    size_t po = (size_t)(row - pre_row0) * DMODEL + h * HD + hd0;
    pre[po] = f2bf(x0); pre[po + 1] = f2bf(x1);
  }
}

// V fp32 [S][24*128] -> Vt bf16 [24][128][S]  (so P*V B-frags load contiguous)
__global__ __launch_bounds__(256) void v_transpose(
    const float* __restrict__ v, __bf16* __restrict__ vt, int S) {
  size_t idx = (size_t)blockIdx.x * blockDim.x + threadIdx.x;
  size_t total = (size_t)S * DMODEL;
  if (idx >= total) return;
  int s = (int)(idx % S);
  int o = (int)(idx / S);  // o = h*128 + hd
  vt[idx] = f2bf(v[(size_t)s * DMODEL + o]);
}

__global__ __launch_bounds__(256) void cvt_bf(
    const float* __restrict__ src, __bf16* __restrict__ dst, size_t n) {
  size_t i = (size_t)blockIdx.x * blockDim.x + threadIdx.x;
  if (i < n) dst[i] = f2bf(src[i]);
}

__global__ __launch_bounds__(256) void combine_img(
    const float* __restrict__ hs, const float* __restrict__ iphs,
    float* __restrict__ out, size_t n) {
  size_t i = (size_t)blockIdx.x * blockDim.x + threadIdx.x;
  if (i < n) out[i] = hs[(size_t)STXT * DMODEL + i] + iphs[i];  // IP_SCALE = 1
}

// ---------------------------------------------------------------------------
// Flash attention (one wave = 16 query rows x full 128 head dim).
// Q,K bf16 [S][24*128]; Vt bf16 [24][128][Skv]; O fp32 [Sq][24*128].
// 32-key chunks: preload 8 K-frags then a fenced 8-WMMA burst for S = Q*K^T,
// online softmax (shfl within 16-lane halves, matching the D-layout), P goes
// D-layout -> LDS -> A-layout, then 8 WMMAs (paired V-frags) for P*V.
// block = 4 waves (4 q-tiles, same head).
// ---------------------------------------------------------------------------
__global__ __launch_bounds__(128) void attn_kernel(
    const __bf16* __restrict__ Q, const __bf16* __restrict__ K,
    const __bf16* __restrict__ Vt, float* __restrict__ O,
    int Sq, int Skv, float scale) {
  __shared__ __bf16 plds[4][16][40];  // +pad against bank conflicts
  int wave = threadIdx.x >> 5, lane = threadIdx.x & 31;
  int h = blockIdx.y;
  int q0 = blockIdx.x * 64 + wave * 16;
  const __bf16* Qh  = Q + h * HD;
  const __bf16* Kh  = K + h * HD;
  const __bf16* Vth = Vt + (size_t)h * HD * Skv;

  v16bf qa[4];
#pragma unroll
  for (int kk = 0; kk < 4; ++kk) qa[kk] = load_frag_bf16(Qh, DMODEL, q0, kk * 32);

  v8f acc[8];
  float mrow[8], lrow[8];
#pragma unroll
  for (int j = 0; j < 8; ++j)
#pragma unroll
    for (int i = 0; i < 8; ++i) acc[j][i] = 0.0f;
#pragma unroll
  for (int i = 0; i < 8; ++i) { mrow[i] = -3.0e38f; lrow[i] = 0.0f; }

  for (int c0 = 0; c0 < Skv; c0 += 32) {
    // preload the whole 32x128 K chunk as 8 independent fragments, then burst
    v16bf kb[8];
#pragma unroll
    for (int kk = 0; kk < 4; ++kk) {
      kb[2 * kk]     = load_frag_bf16(Kh, DMODEL, c0,      kk * 32);
      kb[2 * kk + 1] = load_frag_bf16(Kh, DMODEL, c0 + 16, kk * 32);
    }
    v8f s0, s1;
#pragma unroll
    for (int i = 0; i < 8; ++i) { s0[i] = 0.0f; s1[i] = 0.0f; }
    SCHED_FENCE();
#pragma unroll
    for (int kk = 0; kk < 4; ++kk) {
      s0 = wmma_bf16(qa[kk], kb[2 * kk],     s0);
      s1 = wmma_bf16(qa[kk], kb[2 * kk + 1], s1);
    }
    SCHED_FENCE();
    float corr[8];
#pragma unroll
    for (int i = 0; i < 8; ++i) {
      float a0 = s0[i] * scale, a1 = s1[i] * scale;
      float mx = fmaxf(a0, a1);
#pragma unroll
      for (int d = 1; d < 16; d <<= 1) mx = fmaxf(mx, __shfl_xor(mx, d, 32));
      float mnew = fmaxf(mrow[i], mx);
      corr[i] = __expf(mrow[i] - mnew);
      float p0 = __expf(a0 - mnew), p1 = __expf(a1 - mnew);
      float ps = p0 + p1;
#pragma unroll
      for (int d = 1; d < 16; d <<= 1) ps += __shfl_xor(ps, d, 32);
      lrow[i] = lrow[i] * corr[i] + ps;
      mrow[i] = mnew;
      s0[i] = p0; s1[i] = p1;
    }
#pragma unroll
    for (int j = 0; j < 8; ++j)
#pragma unroll
      for (int i = 0; i < 8; ++i) acc[j][i] *= corr[i];

    // P: D-layout -> LDS (row m, col n)
    int mh = (lane >> 4) << 3, n = lane & 15;
#pragma unroll
    for (int i = 0; i < 8; ++i) {
      plds[wave][mh + i][n]      = f2bf(s0[i]);
      plds[wave][mh + i][n + 16] = f2bf(s1[i]);
    }
    __syncthreads();
    // read back in A-layout
    int r = lane & 15, cb = (lane >> 4) << 3;
    v16bf pa;
#pragma unroll
    for (int e = 0; e < 8; ++e) {
      pa[e]     = plds[wave][r][cb + e];
      pa[8 + e] = plds[wave][r][cb + 16 + e];
    }
    __syncthreads();
    // P*V with paired V fragments (B[k=kv][n=hd] from Vt, contiguous loads)
#pragma unroll
    for (int j = 0; j < 8; j += 2) {
      v16bf vb0 = load_frag_bf16(Vth, Skv, 16 * j,       c0);
      v16bf vb1 = load_frag_bf16(Vth, Skv, 16 * (j + 1), c0);
      acc[j]     = wmma_bf16(pa, vb0, acc[j]);
      acc[j + 1] = wmma_bf16(pa, vb1, acc[j + 1]);
    }
  }
  int rb = q0 + ((lane >> 4) << 3), cl = lane & 15;
#pragma unroll
  for (int j = 0; j < 8; ++j)
#pragma unroll
    for (int i = 0; i < 8; ++i)
      O[(size_t)(rb + i) * DMODEL + h * HD + 16 * j + cl] = acc[j][i] / lrow[i];
}

// ---------------------------------------------------------------------------
extern "C" void kernel_launch(void* const* d_in, const int* in_sizes, int n_in,
                              void* d_out, int out_size, void* d_ws, size_t ws_size,
                              hipStream_t stream) {
  const float* hid   = (const float*)d_in[0];   // [2048][3072]
  const float* enc   = (const float*)d_in[1];   // [512][3072]
  const float* iemb  = (const float*)d_in[2];   // [64][4096]
  const float* cosb  = (const float*)d_in[3];   // [2560][128]
  const float* sinb  = (const float*)d_in[4];
  const float* Wq  = (const float*)d_in[5];  const float* bq  = (const float*)d_in[6];
  const float* Wk  = (const float*)d_in[7];  const float* bk  = (const float*)d_in[8];
  const float* Wv  = (const float*)d_in[9];  const float* bv  = (const float*)d_in[10];
  const float* Waq = (const float*)d_in[11]; const float* baq = (const float*)d_in[12];
  const float* Wak = (const float*)d_in[13]; const float* bak = (const float*)d_in[14];
  const float* Wav = (const float*)d_in[15]; const float* bav = (const float*)d_in[16];
  const float* Wo  = (const float*)d_in[17]; const float* bo  = (const float*)d_in[18];
  const float* Wao = (const float*)d_in[19]; const float* bao = (const float*)d_in[20];
  const float* nqw  = (const float*)d_in[21];
  const float* nkw  = (const float*)d_in[22];
  const float* naqw = (const float*)d_in[23];
  const float* nakw = (const float*)d_in[24];
  const float* Wkip = (const float*)d_in[25];  // [3072][4096]
  const float* Wvip = (const float*)d_in[26];
  float* out = (float*)d_out;                  // [2048+512][3072]

  // ---- workspace carve (≈227 MB) ----
  char* p = (char*)d_ws;
  auto carve = [&](size_t bytes) -> char* {
    char* r = p; p += (bytes + 255) & ~(size_t)255; return r;
  };
  float*  qf     = (float*) carve((size_t)SALL * DMODEL * 4);
  float*  kf     = (float*) carve((size_t)SALL * DMODEL * 4);
  float*  vf     = (float*) carve((size_t)SALL * DMODEL * 4);
  float*  ipkf   = (float*) carve((size_t)NIP  * DMODEL * 4);
  float*  ipvf   = (float*) carve((size_t)NIP  * DMODEL * 4);
  __bf16* ipk_bf = (__bf16*)carve((size_t)NIP  * DMODEL * 2);
  __bf16* qr_bf  = (__bf16*)carve((size_t)SALL * DMODEL * 2);
  __bf16* kr_bf  = (__bf16*)carve((size_t)SALL * DMODEL * 2);
  __bf16* qpre   = (__bf16*)carve((size_t)SIMG * DMODEL * 2);
  __bf16* Vt     = (__bf16*)carve((size_t)NH * HD * SALL * 2);
  __bf16* Vtip   = (__bf16*)carve((size_t)NH * HD * NIP  * 2);
  float*  hs     = (float*) carve((size_t)SALL * DMODEL * 4);
  float*  iphs   = (float*) carve((size_t)SIMG * DMODEL * 4);
  float*  imgc   = (float*) carve((size_t)SIMG * DMODEL * 4);

  auto gemm = [&](const float* A, const float* W, const float* bias, float* C,
                  int M, int N, int K) {
    dim3 g(N / 64, (M + 127) / 128);
    gemm_xwT<<<g, dim3(128), 0, stream>>>(A, W, bias, C, M, N, K);
  };

  // 1) projections (concat along seq = just row offsets in [s][h*128+hd])
  gemm(enc, Waq, baq, qf,                         STXT, DMODEL, DMODEL);
  gemm(hid, Wq,  bq,  qf + (size_t)STXT * DMODEL, SIMG, DMODEL, DMODEL);
  gemm(enc, Wak, bak, kf,                         STXT, DMODEL, DMODEL);
  gemm(hid, Wk,  bk,  kf + (size_t)STXT * DMODEL, SIMG, DMODEL, DMODEL);
  gemm(enc, Wav, bav, vf,                         STXT, DMODEL, DMODEL);
  gemm(hid, Wv,  bv,  vf + (size_t)STXT * DMODEL, SIMG, DMODEL, DMODEL);
  gemm(iemb, Wkip, nullptr, ipkf, NIP, DMODEL, EMB);
  gemm(iemb, Wvip, nullptr, ipvf, NIP, DMODEL, EMB);

  // 2) per-head RMS norm (txt rows use norm_a*, img rows use norm_*)
  {
    int waves = SALL * NH, blocks = (waves + 7) / 8;
    rms_heads<<<blocks, 256, 0, stream>>>(qf, naqw, nqw, STXT, SALL);
    rms_heads<<<blocks, 256, 0, stream>>>(kf, nakw, nkw, STXT, SALL);
  }

  // 3) RoPE + bf16 convert (q also emits pre-RoPE image rows for IP branch)
  {
    size_t total = (size_t)SALL * NH * (HD / 2);
    int blocks = (int)((total + 255) / 256);
    rope_cvt<<<blocks, 256, 0, stream>>>(qf, cosb, sinb, qr_bf, qpre, SALL, STXT);
    rope_cvt<<<blocks, 256, 0, stream>>>(kf, cosb, sinb, kr_bf, nullptr, SALL, 0);
  }

  // 4) V transposes + ip_k convert
  v_transpose<<<(int)(((size_t)SALL * DMODEL + 255) / 256), 256, 0, stream>>>(vf, Vt, SALL);
  v_transpose<<<(int)(((size_t)NIP  * DMODEL + 255) / 256), 256, 0, stream>>>(ipvf, Vtip, NIP);
  cvt_bf<<<(int)(((size_t)NIP * DMODEL + 255) / 256), 256, 0, stream>>>(
      ipkf, ipk_bf, (size_t)NIP * DMODEL);

  // 5) attention
  const float sm_scale = 0.08838834764831845f;  // 1/sqrt(128)
  attn_kernel<<<dim3(SIMG / 64, NH), dim3(128), 0, stream>>>(
      qpre, ipk_bf, Vtip, iphs, SIMG, NIP, sm_scale);
  attn_kernel<<<dim3(SALL / 64, NH), dim3(128), 0, stream>>>(
      qr_bf, kr_bf, Vt, hs, SALL, SALL, sm_scale);

  // 6) img = hs[512:] + ip_hs  (IP_SCALE = 1)
  combine_img<<<(int)(((size_t)SIMG * DMODEL + 255) / 256), 256, 0, stream>>>(
      hs, iphs, imgc, (size_t)SIMG * DMODEL);

  // 7) output projections straight into d_out (img then enc)
  gemm(imgc, Wo,  bo,  out,                         SIMG, DMODEL, DMODEL);
  gemm(hs,   Wao, bao, out + (size_t)SIMG * DMODEL, STXT, DMODEL, DMODEL);
}
